// LinkPredict_48747878810303
// MI455X (gfx1250) — compile-verified, matrix operand order checked by reading, not run on hardware
//
#include <hip/hip_runtime.h>
#include <hip/hip_bf16.h>

// ---------------------------------------------------------------------------
// CDNA5 / gfx1250 link-prediction pipeline.
// - Heavy GEMMs use V_WMMA_F32_16X16X32_BF16 with bf16 hi/lo split precision
//   (3 WMMAs per 16x16x32 step => ~fp32 accuracy on the matrix pipes).
// - The dominant score GEMM consumes pre-split bf16 operands and stages tiles
//   with a DOUBLE-BUFFERED GLOBAL_LOAD_ASYNC_TO_LDS_B128 pipeline (ASYNCcnt),
//   overlapping the async copies with the WMMA chain. Wave tile 32x16 gives
//   12 WMMAs per 16 ds_load_b128 per K-step.
// ---------------------------------------------------------------------------

typedef __attribute__((ext_vector_type(16))) __bf16         v16bf;
typedef __attribute__((ext_vector_type(8)))  float          v8f;
typedef __attribute__((ext_vector_type(8)))  unsigned short ushort8;

#define HDIM  256
#define NBINS 2048

union FragU { ushort8 u[2]; v16bf v; };

__device__ __forceinline__ void split_bf16(float f, unsigned short& hi, unsigned short& lo) {
    unsigned uh = __float_as_uint(f) & 0xFFFF0000u;   // truncate to bf16 "hi"
    hi = (unsigned short)(uh >> 16);
    float r = f - __uint_as_float(uh);                // residual
    lo = (unsigned short)(__float_as_uint(r) >> 16);
}

// monotone int mapping for float atomicMax via integer atomics
__device__ __forceinline__ unsigned f2mono(float f) {
    unsigned u = __float_as_uint(f);
    return (u & 0x80000000u) ? ~u : (u | 0x80000000u);
}
__device__ __forceinline__ float mono2f(unsigned u) {
    return __uint_as_float((u & 0x80000000u) ? (u & 0x7FFFFFFFu) : ~u);
}

// 32-bit LDS offset for async-to-LDS dest (flat->LDS keeps addr[31:0])
__device__ __forceinline__ unsigned lds_off(const void* p) {
    return (unsigned)(size_t)p;
}

__global__ void zero_kernel(float* p, int n) {
    int i = blockIdx.x * 256 + threadIdx.x;
    if (i < n) p[i] = 0.0f;
}

// one-shot f32 -> bf16 hi/lo split of a whole matrix
__global__ void split_mat_kernel(const float* __restrict__ X, unsigned short* __restrict__ hi,
                                 unsigned short* __restrict__ lo, int n) {
    int i = blockIdx.x * 256 + threadIdx.x;
    if (i < n) {
        unsigned short h, l;
        split_bf16(X[i], h, l);
        hi[i] = h; lo[i] = l;
    }
}

// ---------------------------------------------------------------------------
// Generic C[M,N] = A[M,K]*B[N,K]^T (fp32 in/out, in-kernel split). Used for
// the small GEMMs (h = enc@W^T, proj = tail@Wi^T): ~11% of total FLOPs.
// block tile 64(M) x 32(N), 8 waves (4x2), K stepped by 32.
// ---------------------------------------------------------------------------
__global__ void gemm_nt_kernel(const float* __restrict__ A, const float* __restrict__ B,
                               float* __restrict__ C, int M, int N, int K) {
    __shared__ __align__(16) unsigned short lAhi[64][32], lAlo[64][32];
    __shared__ __align__(16) unsigned short lBhi[32][32], lBlo[32][32];

    const int tid  = threadIdx.x;
    const int lane = tid & 31, wid = tid >> 5;
    const int wm = wid >> 1, wn = wid & 1;
    const int bM = blockIdx.x * 64, bN = blockIdx.y * 32;
    const int hi8 = (lane >> 4) << 3;   // 0 or 8: K-half select (A/B), M-half (C)
    const int l15 = lane & 15;

    v8f c = {};
    const int ar = tid >> 2, ac = (tid & 3) * 8;

    for (int k0 = 0; k0 < K; k0 += 32) {
        __syncthreads();
        {
            const float* ap = A + (size_t)(bM + ar) * K + k0 + ac;
            float4 v0 = *(const float4*)ap, v1 = *(const float4*)(ap + 4);
            __builtin_prefetch(ap + 32, 0, 1);
            float vv[8] = {v0.x, v0.y, v0.z, v0.w, v1.x, v1.y, v1.z, v1.w};
#pragma unroll
            for (int i = 0; i < 8; ++i) split_bf16(vv[i], lAhi[ar][ac + i], lAlo[ar][ac + i]);
        }
        if (tid < 128) {
            const int br = tid >> 2, bc = (tid & 3) * 8;
            const float* bp = B + (size_t)(bN + br) * K + k0 + bc;
            float4 v0 = *(const float4*)bp, v1 = *(const float4*)(bp + 4);
            __builtin_prefetch(bp + 32, 0, 1);
            float vv[8] = {v0.x, v0.y, v0.z, v0.w, v1.x, v1.y, v1.z, v1.w};
#pragma unroll
            for (int i = 0; i < 8; ++i) split_bf16(vv[i], lBhi[br][bc + i], lBlo[br][bc + i]);
        }
        __syncthreads();

        FragU ah, al, bh, bl;
        const int am = wm * 16 + l15;
        const int bn = wn * 16 + l15;
        ah.u[0] = *(const ushort8*)&lAhi[am][hi8];  ah.u[1] = *(const ushort8*)&lAhi[am][hi8 + 16];
        al.u[0] = *(const ushort8*)&lAlo[am][hi8];  al.u[1] = *(const ushort8*)&lAlo[am][hi8 + 16];
        bh.u[0] = *(const ushort8*)&lBhi[bn][hi8];  bh.u[1] = *(const ushort8*)&lBhi[bn][hi8 + 16];
        bl.u[0] = *(const ushort8*)&lBlo[bn][hi8];  bl.u[1] = *(const ushort8*)&lBlo[bn][hi8 + 16];

        c = __builtin_amdgcn_wmma_f32_16x16x32_bf16(false, ah.v, false, bh.v, (short)0, c, false, false);
        c = __builtin_amdgcn_wmma_f32_16x16x32_bf16(false, ah.v, false, bl.v, (short)0, c, false, false);
        c = __builtin_amdgcn_wmma_f32_16x16x32_bf16(false, al.v, false, bh.v, (short)0, c, false, false);
    }
#pragma unroll
    for (int r = 0; r < 8; ++r) {
        int row = bM + wm * 16 + r + hi8;
        int col = bN + wn * 16 + l15;
        C[(size_t)row * N + col] = c[r];
    }
}

// ---------------------------------------------------------------------------
// Dominant fused score GEMM: s0 = A*B0^T, s1 = A*B1^T sharing the A tile.
// Block tile 128(M) x 32(N), 8 waves (4x2), wave tile 32x16.
// Double-buffered async-to-LDS staging overlaps copies with WMMA.
// Epilogue writes score_1, adj mask, and a masked atomic histogram.
// ---------------------------------------------------------------------------
__global__ void gemm_score_kernel(const unsigned short* __restrict__ Ahi,
                                  const unsigned short* __restrict__ Alo,
                                  const unsigned short* __restrict__ B0hi,
                                  const unsigned short* __restrict__ B0lo,
                                  const unsigned short* __restrict__ B1hi,
                                  const unsigned short* __restrict__ B1lo,
                                  float* __restrict__ score,
                                  unsigned char* __restrict__ adj,
                                  unsigned* __restrict__ hist,
                                  int N, int K) {
    __shared__ __align__(16) unsigned short lAhi[2][128][32], lAlo[2][128][32];
    __shared__ __align__(16) unsigned short lB0hi[2][32][32], lB0lo[2][32][32];
    __shared__ __align__(16) unsigned short lB1hi[2][32][32], lB1lo[2][32][32];

    const int tid  = threadIdx.x;
    const int lane = tid & 31, wid = tid >> 5;
    const int wm = wid >> 1, wn = wid & 1;              // 4 (M) x 2 (N) waves
    const int bM = blockIdx.x * 128, bN = blockIdx.y * 32;
    const int hi8 = (lane >> 4) << 3;
    const int l15 = lane & 15;

    // staging geometry: A tile 128x32 -> 2x16B chunks/thread; B tiles 32x32 -> 1 chunk
    const int arow = tid >> 1, acol = (tid & 1) * 16;   // chunk pair at acol, acol+8
    const int t128 = tid & 127;
    const int brow = t128 >> 2, bcol = (t128 & 3) * 8;
    const bool firstB = (tid < 128);

    const unsigned ldsAh[2] = { lds_off(&lAhi[0][arow][acol]), lds_off(&lAhi[1][arow][acol]) };
    const unsigned ldsAl[2] = { lds_off(&lAlo[0][arow][acol]), lds_off(&lAlo[1][arow][acol]) };
    const unsigned ldsBh[2] = { firstB ? lds_off(&lB0hi[0][brow][bcol]) : lds_off(&lB1hi[0][brow][bcol]),
                                firstB ? lds_off(&lB0hi[1][brow][bcol]) : lds_off(&lB1hi[1][brow][bcol]) };
    const unsigned ldsBl[2] = { firstB ? lds_off(&lB0lo[0][brow][bcol]) : lds_off(&lB1lo[0][brow][bcol]),
                                firstB ? lds_off(&lB0lo[1][brow][bcol]) : lds_off(&lB1lo[1][brow][bcol]) };

    const unsigned short* gAh = Ahi + (size_t)(bM + arow) * K + acol;
    const unsigned short* gAl = Alo + (size_t)(bM + arow) * K + acol;
    const unsigned short* gBh = (firstB ? B0hi : B1hi) + (size_t)(bN + brow) * K + bcol;
    const unsigned short* gBl = (firstB ? B0lo : B1lo) + (size_t)(bN + brow) * K + bcol;

    // async-stage one K-step tile set into LDS buffer `buf`
    auto stage = [&](int buf, int k0) {
        asm volatile(
            "global_load_async_to_lds_b128 %0, %4, off\n\t"
            "global_load_async_to_lds_b128 %0, %4, off offset:16\n\t"
            "global_load_async_to_lds_b128 %1, %5, off\n\t"
            "global_load_async_to_lds_b128 %1, %5, off offset:16\n\t"
            "global_load_async_to_lds_b128 %2, %6, off\n\t"
            "global_load_async_to_lds_b128 %3, %7, off"
            :: "v"(ldsAh[buf]), "v"(ldsAl[buf]), "v"(ldsBh[buf]), "v"(ldsBl[buf]),
               "v"((unsigned long long)(size_t)(gAh + k0)),
               "v"((unsigned long long)(size_t)(gAl + k0)),
               "v"((unsigned long long)(size_t)(gBh + k0)),
               "v"((unsigned long long)(size_t)(gBl + k0))
            : "memory");
    };

    v8f c0a = {}, c0b = {}, c1a = {}, c1b = {};

    stage(0, 0);   // prologue fill of buffer 0
    for (int k0 = 0; k0 < K; k0 += 32) {
        const int cur = (k0 >> 5) & 1;
        asm volatile("s_wait_asynccnt 0x0" ::: "memory");  // my copies for `cur` done
        __syncthreads();                                   // everyone's copies visible;
                                                           // prior reads of cur^1 drained
        if (k0 + 32 < K) stage(cur ^ 1, k0 + 32);          // prefetch next under compute

        FragU ahA, alA, ahB, alB, b0h, b0l, b1h, b1l;
        const int amA = wm * 32 + l15, amB = amA + 16;
        const int bn  = wn * 16 + l15;
        ahA.u[0] = *(const ushort8*)&lAhi[cur][amA][hi8];  ahA.u[1] = *(const ushort8*)&lAhi[cur][amA][hi8 + 16];
        alA.u[0] = *(const ushort8*)&lAlo[cur][amA][hi8];  alA.u[1] = *(const ushort8*)&lAlo[cur][amA][hi8 + 16];
        ahB.u[0] = *(const ushort8*)&lAhi[cur][amB][hi8];  ahB.u[1] = *(const ushort8*)&lAhi[cur][amB][hi8 + 16];
        alB.u[0] = *(const ushort8*)&lAlo[cur][amB][hi8];  alB.u[1] = *(const ushort8*)&lAlo[cur][amB][hi8 + 16];
        b0h.u[0] = *(const ushort8*)&lB0hi[cur][bn][hi8];  b0h.u[1] = *(const ushort8*)&lB0hi[cur][bn][hi8 + 16];
        b0l.u[0] = *(const ushort8*)&lB0lo[cur][bn][hi8];  b0l.u[1] = *(const ushort8*)&lB0lo[cur][bn][hi8 + 16];
        b1h.u[0] = *(const ushort8*)&lB1hi[cur][bn][hi8];  b1h.u[1] = *(const ushort8*)&lB1hi[cur][bn][hi8 + 16];
        b1l.u[0] = *(const ushort8*)&lB1lo[cur][bn][hi8];  b1l.u[1] = *(const ushort8*)&lB1lo[cur][bn][hi8 + 16];

        c0a = __builtin_amdgcn_wmma_f32_16x16x32_bf16(false, ahA.v, false, b0h.v, (short)0, c0a, false, false);
        c0a = __builtin_amdgcn_wmma_f32_16x16x32_bf16(false, ahA.v, false, b0l.v, (short)0, c0a, false, false);
        c0a = __builtin_amdgcn_wmma_f32_16x16x32_bf16(false, alA.v, false, b0h.v, (short)0, c0a, false, false);
        c1a = __builtin_amdgcn_wmma_f32_16x16x32_bf16(false, ahA.v, false, b1h.v, (short)0, c1a, false, false);
        c1a = __builtin_amdgcn_wmma_f32_16x16x32_bf16(false, ahA.v, false, b1l.v, (short)0, c1a, false, false);
        c1a = __builtin_amdgcn_wmma_f32_16x16x32_bf16(false, alA.v, false, b1h.v, (short)0, c1a, false, false);
        c0b = __builtin_amdgcn_wmma_f32_16x16x32_bf16(false, ahB.v, false, b0h.v, (short)0, c0b, false, false);
        c0b = __builtin_amdgcn_wmma_f32_16x16x32_bf16(false, ahB.v, false, b0l.v, (short)0, c0b, false, false);
        c0b = __builtin_amdgcn_wmma_f32_16x16x32_bf16(false, alB.v, false, b0h.v, (short)0, c0b, false, false);
        c1b = __builtin_amdgcn_wmma_f32_16x16x32_bf16(false, ahB.v, false, b1h.v, (short)0, c1b, false, false);
        c1b = __builtin_amdgcn_wmma_f32_16x16x32_bf16(false, ahB.v, false, b1l.v, (short)0, c1b, false, false);
        c1b = __builtin_amdgcn_wmma_f32_16x16x32_bf16(false, alB.v, false, b1h.v, (short)0, c1b, false, false);
    }

    auto emit = [&](int row, int col, float s0, float s1) {
        size_t idx = (size_t)row * N + col;
        score[idx] = s1;
        unsigned char a = (s1 > s0) ? 1 : 0;
        adj[idx] = a;
        if (a) {
            int b = (int)((s1 + 64.0f) * (NBINS / 128.0f));
            b = b < 0 ? 0 : (b > NBINS - 1 ? NBINS - 1 : b);
            atomicAdd(&hist[b], 1u);
        }
    };
    const int col = bN + wn * 16 + l15;
#pragma unroll
    for (int r = 0; r < 8; ++r) emit(bM + wm * 32 + r + hi8,      col, c0a[r], c1a[r]);
#pragma unroll
    for (int r = 0; r < 8; ++r) emit(bM + wm * 32 + 16 + r + hi8, col, c0b[r], c1b[r]);
}

// ---------------------------------------------------------------------------
// GAT helpers
// ---------------------------------------------------------------------------
__global__ void rowdot_kernel(const float* __restrict__ h, const float* __restrict__ asv,
                              const float* __restrict__ adv, float* __restrict__ as_,
                              float* __restrict__ ad_) {
    const int wid = threadIdx.x >> 5, lane = threadIdx.x & 31;
    const int row = blockIdx.x * 8 + wid;
    const float* hr = h + (size_t)row * HDIM;
    float s0 = 0.f, s1 = 0.f;
    for (int c = lane; c < HDIM; c += 32) { float v = hr[c]; s0 += v * asv[c]; s1 += v * adv[c]; }
    for (int o = 16; o; o >>= 1) { s0 += __shfl_xor(s0, o, 32); s1 += __shfl_xor(s1, o, 32); }
    if (lane == 0) { as_[row] = s0; ad_[row] = s1; }
}

__global__ void edge_e_kernel(const int* __restrict__ ei, const float* __restrict__ as_,
                              const float* __restrict__ ad_, float* __restrict__ ew,
                              unsigned* __restrict__ emax, int E, int Nn) {
    int t = blockIdx.x * 256 + threadIdx.x;
    if (t >= E + Nn) return;
    int s = (t < E) ? ei[t]     : (t - E);
    int d = (t < E) ? ei[E + t] : (t - E);
    float e = as_[s] + ad_[d];
    e = e > 0.0f ? e : 0.2f * e;                 // LeakyReLU(0.2)
    ew[t] = e;
    atomicMax(&emax[d], f2mono(e));
}

__global__ void edge_w_kernel(const int* __restrict__ ei, float* __restrict__ ew,
                              const unsigned* __restrict__ emax, float* __restrict__ denom,
                              int E, int Nn) {
    int t = blockIdx.x * 256 + threadIdx.x;
    if (t >= E + Nn) return;
    int d = (t < E) ? ei[E + t] : (t - E);
    float w = __expf(ew[t] - mono2f(emax[d]));
    ew[t] = w;
    atomicAdd(&denom[d], w);
}

__global__ void edge_agg_kernel(const int* __restrict__ ei, const float* __restrict__ h,
                                const float* __restrict__ ew, const float* __restrict__ denom,
                                float* __restrict__ out, int E) {
    int t = blockIdx.x;
    int s = (t < E) ? ei[t]     : (t - E);
    int d = (t < E) ? ei[E + t] : (t - E);
    float alpha = ew[t] / denom[d];
    int c = threadIdx.x;
    atomicAdd(&out[(size_t)d * HDIM + c], alpha * h[(size_t)s * HDIM + c]);
}

__global__ void bias_kernel(float* __restrict__ out, const float* __restrict__ b) {
    out[(size_t)blockIdx.x * HDIM + threadIdx.x] += b[threadIdx.x];
}

// ---------------------------------------------------------------------------
// Selection: histogram threshold, then streaming compaction capped at kcap.
// ---------------------------------------------------------------------------
__global__ void threshold_kernel(const unsigned* __restrict__ hist, float* __restrict__ thr,
                                 int kcap) {
    unsigned cum = 0;
    float t = -3.0e38f;
    for (int b = NBINS - 1; b >= 0; --b) {
        cum += hist[b];
        if (cum >= (unsigned)kcap) { t = -64.0f + b * (128.0f / NBINS); break; }
    }
    thr[0] = t;
}

__global__ void compact_kernel(const float* __restrict__ score, const unsigned char* __restrict__ adj,
                               const float* __restrict__ thr, unsigned* __restrict__ counter,
                               float* __restrict__ out, int kcap, int N) {
    int idx = blockIdx.x * 256 + threadIdx.x;
    float t = thr[0];
    float s = score[idx];
    if (adj[idx] && s >= t) {
        unsigned pos = atomicAdd(counter, 1u);
        if (pos < (unsigned)kcap) {
            int i = idx / N, j = idx - i * N;
            out[pos]            = (float)i;
            out[kcap + pos]     = (float)j;
            out[2 * kcap + pos] = 0.5f + 0.5f / (1.0f + __expf(-s));  // (sigmoid+1)/2
        }
    }
}

// ---------------------------------------------------------------------------
extern "C" void kernel_launch(void* const* d_in, const int* in_sizes, int n_in,
                              void* d_out, int out_size, void* d_ws, size_t ws_size,
                              hipStream_t stream) {
    const float* enc        = (const float*)d_in[0];
    const int*   ei         = (const int*)  d_in[1];
    const float* W_head     = (const float*)d_in[2];
    const float* a_src_head = (const float*)d_in[3];
    const float* a_dst_head = (const float*)d_in[4];
    const float* b_head     = (const float*)d_in[5];
    const float* W_tail     = (const float*)d_in[6];
    const float* a_src_tail = (const float*)d_in[7];
    const float* a_dst_tail = (const float*)d_in[8];
    const float* b_tail     = (const float*)d_in[9];
    const float* W0         = (const float*)d_in[10];
    const float* W1         = (const float*)d_in[11];

    const int N    = in_sizes[0] / HDIM;   // 4096
    const int E    = in_sizes[1] / 2;      // 131072
    const int EN   = E + N;
    const int kcap = out_size / 3;

    // workspace layout (4-byte granules)
    float* ws       = (float*)d_ws;
    const size_t HB = (size_t)N * HDIM;
    float*    f_h     = ws;
    float*    f_head  = ws + 1 * HB;
    float*    f_tail  = ws + 2 * HB;
    float*    f_p0    = ws + 3 * HB;
    float*    f_p1    = ws + 4 * HB;
    float*    f_as    = ws + 5 * HB;
    float*    f_ad    = f_as + N;
    unsigned* u_emax  = (unsigned*)(f_ad + N);
    float*    f_denom = (float*)(u_emax + N);
    float*    f_ew    = f_denom + N;                         // EN floats
    unsigned* u_hist  = (unsigned*)(f_ew + EN + 64);         // NBINS bins + counter
    unsigned* u_cnt   = u_hist + NBINS;
    float*    f_thr   = (float*)(u_cnt + 16);
    float*    f_score = f_thr + 64;                          // N*N floats
    unsigned char* adj = (unsigned char*)(f_score + (size_t)N * N);
    // pre-split bf16 hi/lo operand buffers (HB ushorts each; 16B aligned)
    unsigned short* bAhi = (unsigned short*)(adj + (size_t)N * N);
    unsigned short* bAlo = bAhi + HB;
    unsigned short* bP0h = bAlo + HB;
    unsigned short* bP0l = bP0h + HB;
    unsigned short* bP1h = bP0l + HB;
    unsigned short* bP1l = bP1h + HB;

    auto Z = [&](float* p, size_t cnt) {
        zero_kernel<<<dim3((unsigned)((cnt + 255) / 256)), 256, 0, stream>>>(p, (int)cnt);
    };

    // per-call zeroing of accumulation targets
    Z(f_head, HB);
    Z(f_tail, HB);
    Z((float*)u_hist, NBINS + 16);
    Z((float*)d_out, (size_t)out_size);

    auto gat = [&](const float* W, const float* asv, const float* adv,
                   const float* bv, float* outp) {
        Z((float*)u_emax, N);
        Z(f_denom, N);
        gemm_nt_kernel<<<dim3(N / 64, HDIM / 32), 256, 0, stream>>>(enc, W, f_h, N, HDIM, HDIM);
        rowdot_kernel<<<dim3(N / 8), 256, 0, stream>>>(f_h, asv, adv, f_as, f_ad);
        edge_e_kernel<<<dim3((EN + 255) / 256), 256, 0, stream>>>(ei, f_as, f_ad, f_ew, u_emax, E, N);
        edge_w_kernel<<<dim3((EN + 255) / 256), 256, 0, stream>>>(ei, f_ew, u_emax, f_denom, E, N);
        edge_agg_kernel<<<dim3(EN), 256, 0, stream>>>(ei, f_h, f_ew, f_denom, outp, E);
        bias_kernel<<<dim3(N), 256, 0, stream>>>(outp, bv);
    };
    gat(W_head, a_src_head, a_dst_head, b_head, f_head);
    gat(W_tail, a_src_tail, a_dst_tail, b_tail, f_tail);

    // projections
    gemm_nt_kernel<<<dim3(N / 64, HDIM / 32), 256, 0, stream>>>(f_tail, W0, f_p0, N, HDIM, HDIM);
    gemm_nt_kernel<<<dim3(N / 64, HDIM / 32), 256, 0, stream>>>(f_tail, W1, f_p1, N, HDIM, HDIM);

    // one-shot bf16 hi/lo split of the score-GEMM operands
    const unsigned splitg = (unsigned)((HB + 255) / 256);
    split_mat_kernel<<<splitg, 256, 0, stream>>>(f_head, bAhi, bAlo, (int)HB);
    split_mat_kernel<<<splitg, 256, 0, stream>>>(f_p0,   bP0h, bP0l, (int)HB);
    split_mat_kernel<<<splitg, 256, 0, stream>>>(f_p1,   bP1h, bP1l, (int)HB);

    // fused dual score GEMM (double-buffered async staging) + mask + histogram
    gemm_score_kernel<<<dim3(N / 128, N / 32), 256, 0, stream>>>(
        bAhi, bAlo, bP0h, bP0l, bP1h, bP1l, f_score, adj, u_hist, N, HDIM);

    // threshold + compaction
    threshold_kernel<<<1, 1, 0, stream>>>(u_hist, f_thr, kcap);
    compact_kernel<<<dim3((unsigned)(((size_t)N * N) / 256)), 256, 0, stream>>>(
        f_score, adj, f_thr, u_cnt, (float*)d_out, kcap, N);
}
